// VectorQuantizerEMA_multi_head_56813827391973
// MI455X (gfx1250) — compile-verified, matrix operand order checked by reading, not run on hardware
//
#include <hip/hip_runtime.h>
#include <hip/hip_bf16.h>

typedef __attribute__((ext_vector_type(16))) __bf16   v16bf;
typedef __attribute__((ext_vector_type(8)))  float    v8f;
typedef __attribute__((ext_vector_type(8)))  unsigned v8u;

#define H_   4
#define HD_  128
#define K_   4096
#define N_   16384
#define D_   512
#define QSIZE 8388608ull          // N_*D_
#define GROUPS 256                // K_/16
#define GBYTES 4096               // bytes per code-group of B frags: 4 chunks * 32 lanes * 32B

static __device__ __forceinline__ unsigned pk2bf(float a, float b) {
    unsigned ua = __float_as_uint(a), ub = __float_as_uint(b);
    ua = (ua + 0x7FFFu + ((ua >> 16) & 1u)) >> 16;
    ub = (ub + 0x7FFFu + ((ub >> 16) & 1u)) >> 16;
    return ua | (ub << 16);
}

static __device__ __forceinline__ v16bf as_bf16x16(v8u u) {
    union { v8u u; v16bf b; } c; c.u = u; return c.b;
}

// CDNA5 async copy: global -> LDS (tracked by ASYNCcnt), 16B per lane.
static __device__ __forceinline__ void async_b128(unsigned lds_addr, const void* gaddr) {
    asm volatile("global_load_async_to_lds_b128 %0, %1, off"
                 :: "v"(lds_addr), "v"(gaddr) : "memory");
}
static __device__ __forceinline__ void wait_async0() {
    asm volatile("s_wait_asynccnt 0x0" ::: "memory");
}

// ---------------- pass 1a: enorm[h][k] = sum_d embed[h][d][k]^2 ----------------
__global__ void vq_enorm(const float* __restrict__ embed, float* __restrict__ enorm) {
    int t = blockIdx.x * blockDim.x + threadIdx.x;     // [0, H_*K_)
    int h = t >> 12, k = t & (K_ - 1);
    const float* E = embed + (size_t)h * HD_ * K_;
    float s = 0.f;
    #pragma unroll 4
    for (int d = 0; d < HD_; ++d) { float v = E[(size_t)d * K_ + k]; s = fmaf(v, v, s); }
    enorm[t] = s;
}

// ---------------- pass 1b: embed_t[h][k][d] = embed[h][d][k] (LDS tiled) ----------------
__global__ void vq_transpose(const float* __restrict__ embed, float* __restrict__ et) {
    __shared__ float tile[32][33];
    int h = blockIdx.z, d0 = blockIdx.y * 32, k0 = blockIdx.x * 32;
    const float* E = embed + (size_t)h * HD_ * K_;
    #pragma unroll
    for (int i = 0; i < 4; ++i)
        tile[threadIdx.y + i * 8][threadIdx.x] =
            E[(size_t)(d0 + threadIdx.y + i * 8) * K_ + k0 + threadIdx.x];
    __syncthreads();
    float* T = et + (size_t)h * K_ * HD_;
    #pragma unroll
    for (int i = 0; i < 4; ++i)
        T[(size_t)(k0 + threadIdx.y + i * 8) * HD_ + d0 + threadIdx.x] =
            tile[threadIdx.x][threadIdx.y + i * 8];
}

// ---------------- pass 1c: pack B fragments (WMMA 16x16x32 bf16 lane layout) ----------------
// layout: [h][g][c][lane][8 dwords] -> per (h,g) a contiguous 4KB block.
__global__ void vq_bfrag(const float* __restrict__ embed, unsigned* __restrict__ bfrag) {
    int t = blockIdx.x * blockDim.x + threadIdx.x;     // [0, H_*GROUPS*4*32)
    int lane = t & 31, c = (t >> 5) & 3, g = (t >> 7) & (GROUPS - 1), h = t >> 15;
    int n = g * 16 + (lane & 15);
    int dbase = c * 32 + ((lane >> 4) ? 8 : 0);
    const float* E = embed + (size_t)h * HD_ * K_;
    unsigned o[8];
    #pragma unroll
    for (int v = 0; v < 4; ++v) {
        int d = dbase + 2 * v;
        o[v]     = pk2bf(E[(size_t)d * K_ + n],        E[(size_t)(d + 1) * K_ + n]);
        o[v + 4] = pk2bf(E[(size_t)(d + 16) * K_ + n], E[(size_t)(d + 17) * K_ + n]);
    }
    unsigned* dst = bfrag + (size_t)t * 8;
    #pragma unroll
    for (int v = 0; v < 8; ++v) dst[v] = o[v];
}

// ---------------- pass 2: fused GEMM(argmin) + gather + L1 partial ----------------
// 4 waves / WG share each 4KB B group via async-to-LDS double buffering.
__global__ __launch_bounds__(128) void vq_main(
    const float* __restrict__ x, const unsigned char* __restrict__ bfragB,
    const float* __restrict__ enorm, const float* __restrict__ et,
    float* __restrict__ quant, float* __restrict__ oidx, float* __restrict__ partial) {

    __shared__ uint4 bbuf[2][256];                     // 2 x 4KB

    const int t = threadIdx.x;
    const int wid = t >> 5, lane = t & 31;
    const int h = blockIdx.y;
    const int n0 = blockIdx.x * 128 + wid * 32;        // this wave: 2 tiles of 16 tokens
    const int dof = (lane >> 4) ? 8 : 0;
    const int lm = lane & 15;

    const unsigned char* bfh = bfragB + (size_t)h * GROUPS * GBYTES;
    const unsigned lds0 = (unsigned)(size_t)&bbuf[0][0];
    const unsigned myoff = (unsigned)t * 32u;

    // prologue: start async copy of group 0 into buffer 0
    async_b128(lds0 + myoff,       bfh + myoff);
    async_b128(lds0 + myoff + 16,  bfh + myoff + 16);

    // ---- build A fragments for both 16-token tiles ----
    v16bf a0[4], a1[4];
    {
        const float* xr0 = x + (size_t)(n0 + lm) * D_ + h * HD_;
        const float* xr1 = x + (size_t)(n0 + 16 + lm) * D_ + h * HD_;
        #pragma unroll
        for (int c = 0; c < 4; ++c) {
            const float* p;
            v8u ua;
            p = xr0 + c * 32 + dof;
            {
                float4 f0 = *(const float4*)(p);      float4 f1 = *(const float4*)(p + 4);
                float4 f2 = *(const float4*)(p + 16); float4 f3 = *(const float4*)(p + 20);
                ua[0] = pk2bf(f0.x, f0.y); ua[1] = pk2bf(f0.z, f0.w);
                ua[2] = pk2bf(f1.x, f1.y); ua[3] = pk2bf(f1.z, f1.w);
                ua[4] = pk2bf(f2.x, f2.y); ua[5] = pk2bf(f2.z, f2.w);
                ua[6] = pk2bf(f3.x, f3.y); ua[7] = pk2bf(f3.z, f3.w);
            }
            a0[c] = as_bf16x16(ua);
            p = xr1 + c * 32 + dof;
            {
                float4 f0 = *(const float4*)(p);      float4 f1 = *(const float4*)(p + 4);
                float4 f2 = *(const float4*)(p + 16); float4 f3 = *(const float4*)(p + 20);
                ua[0] = pk2bf(f0.x, f0.y); ua[1] = pk2bf(f0.z, f0.w);
                ua[2] = pk2bf(f1.x, f1.y); ua[3] = pk2bf(f1.z, f1.w);
                ua[4] = pk2bf(f2.x, f2.y); ua[5] = pk2bf(f2.z, f2.w);
                ua[6] = pk2bf(f3.x, f3.y); ua[7] = pk2bf(f3.z, f3.w);
            }
            a1[c] = as_bf16x16(ua);
        }
    }

    float best[16]; int bidx[16];
    #pragma unroll
    for (int j = 0; j < 16; ++j) { best[j] = 3.4e38f; bidx[j] = 0; }

    const float* enh = enorm + h * K_;

    for (int g = 0; g < GROUPS; ++g) {
        const int p = g & 1;
        wait_async0();                 // my 16B x2 for buffer p have landed
        __syncthreads();               // everyone's landed; prior reads of buf[1-p] done
        if (g + 1 < GROUPS) {          // kick off next group into the other buffer
            const unsigned char* src = bfh + (size_t)(g + 1) * GBYTES + myoff;
            const unsigned dst = lds0 + (unsigned)(1 - p) * GBYTES + myoff;
            async_b128(dst,      src);
            async_b128(dst + 16, src + 16);
        }

        v8f acc0 = {0.f,0.f,0.f,0.f,0.f,0.f,0.f,0.f};
        v8f acc1 = {0.f,0.f,0.f,0.f,0.f,0.f,0.f,0.f};
        #pragma unroll
        for (int c = 0; c < 4; ++c) {
            uint4 lo = bbuf[p][c * 64 + lane * 2];
            uint4 hi = bbuf[p][c * 64 + lane * 2 + 1];
            v8u ub; ub[0]=lo.x; ub[1]=lo.y; ub[2]=lo.z; ub[3]=lo.w;
                    ub[4]=hi.x; ub[5]=hi.y; ub[6]=hi.z; ub[7]=hi.w;
            v16bf b = as_bf16x16(ub);
            acc0 = __builtin_amdgcn_wmma_f32_16x16x32_bf16(false, a0[c], false, b,
                                                           (short)0, acc0, false, false);
            acc1 = __builtin_amdgcn_wmma_f32_16x16x32_bf16(false, a1[c], false, b,
                                                           (short)0, acc1, false, false);
        }
        const int code = g * 16 + lm;
        const float en = enh[code];
        #pragma unroll
        for (int j = 0; j < 8; ++j) {
            float s0 = fmaf(-2.f, acc0[j], en);
            float s1 = fmaf(-2.f, acc1[j], en);
            if (s0 < best[j])     { best[j]     = s0; bidx[j]     = code; }
            if (s1 < best[8 + j]) { best[8 + j] = s1; bidx[8 + j] = code; }
        }
    }

    // ---- cross-lane argmin within each 16-lane half (ties -> smaller index) ----
    #pragma unroll
    for (int m = 1; m < 16; m <<= 1) {
        #pragma unroll
        for (int j = 0; j < 16; ++j) {
            float ob = __shfl_xor(best[j], m, 32);
            int   oi = __shfl_xor(bidx[j], m, 32);
            if (ob < best[j] || (ob == best[j] && oi < bidx[j])) { best[j] = ob; bidx[j] = oi; }
        }
    }
    // lanes 0..15 slot j: token (tile)+j ; lanes 16..31 slot j: token (tile)+8+j

    // ---- index output (as float), [N, H] flat ----
    if (lane == 0) {
        #pragma unroll
        for (int j = 0; j < 8; ++j) {
            oidx[(size_t)(n0 + j) * H_ + h]      = (float)bidx[j];
            oidx[(size_t)(n0 + 16 + j) * H_ + h] = (float)bidx[8 + j];
        }
    }
    if (lane == 16) {
        #pragma unroll
        for (int j = 0; j < 8; ++j) {
            oidx[(size_t)(n0 + 8 + j) * H_ + h]  = (float)bidx[j];
            oidx[(size_t)(n0 + 24 + j) * H_ + h] = (float)bidx[8 + j];
        }
    }

    // ---- gather exact f32 code rows, write quantize, accumulate |x - q| ----
    float dsum = 0.f;
    auto emit = [&](int m, int k) {
        const size_t row = (size_t)(n0 + m);
        const float* qr = et + ((size_t)h * K_ + (size_t)k) * HD_ + lane * 4;
        float4 q = *(const float4*)qr;
        const size_t o = row * D_ + h * HD_ + lane * 4;
        float4 xv = *(const float4*)(x + o);
        *(float4*)(quant + o) = q;
        dsum += fabsf(xv.x - q.x) + fabsf(xv.y - q.y) + fabsf(xv.z - q.z) + fabsf(xv.w - q.w);
    };
    #pragma unroll
    for (int jm = 0; jm < 8; ++jm) {
        int k0 = __shfl(bidx[jm], 0, 32);
        int k1 = __shfl(bidx[jm], 16, 32);
        int k2 = __shfl(bidx[8 + jm], 0, 32);
        int k3 = __shfl(bidx[8 + jm], 16, 32);
        emit(jm, k0); emit(8 + jm, k1); emit(16 + jm, k2); emit(24 + jm, k3);
    }

    // ---- wave-reduce L1 partial, one slot per wave (deterministic) ----
    #pragma unroll
    for (int m = 1; m < 32; m <<= 1) dsum += __shfl_xor(dsum, m, 32);
    if (lane == 0)
        partial[((size_t)blockIdx.y * gridDim.x + blockIdx.x) * 4 + wid] = dsum;
}

// ---------------- pass 3: fixed-order reduction -> diff ----------------
__global__ void vq_finalize(const float* __restrict__ partial, float* __restrict__ diffp) {
    __shared__ float s[256];
    float v = 0.f;
    for (int i = threadIdx.x; i < 2048; i += 256) v += partial[i];
    s[threadIdx.x] = v; __syncthreads();
    for (int off = 128; off > 0; off >>= 1) {
        if ((int)threadIdx.x < off) s[threadIdx.x] += s[threadIdx.x + off];
        __syncthreads();
    }
    if (threadIdx.x == 0) *diffp = s[0] * (1.0f / (float)QSIZE);
}

extern "C" void kernel_launch(void* const* d_in, const int* in_sizes, int n_in,
                              void* d_out, int out_size, void* d_ws, size_t ws_size,
                              hipStream_t stream) {
    const float* x     = (const float*)d_in[0];   // [8,2048,512]
    const float* embed = (const float*)d_in[1];   // [4,128,4096]

    float* out   = (float*)d_out;
    float* quant = out;                 // QSIZE floats
    float* diffp = out + QSIZE;         // 1 float
    float* oidx  = out + QSIZE + 1;     // N_*H_ floats

    char* ws = (char*)d_ws;
    unsigned* bfrag = (unsigned*)(ws);                       // 4 MB
    float* enorm    = (float*)(ws + 4194304);                // 64 KB
    float* et       = (float*)(ws + 4259840);                // 8 MB
    float* partial  = (float*)(ws + 12648448);               // 8 KB

    vq_enorm   <<<64, 256, 0, stream>>>(embed, enorm);
    vq_transpose<<<dim3(K_/32, HD_/32, H_), dim3(32, 8), 0, stream>>>(embed, et);
    vq_bfrag   <<<512, 256, 0, stream>>>(embed, bfrag);
    vq_main    <<<dim3(N_/128, H_), 128, 0, stream>>>(x, (const unsigned char*)bfrag,
                                                      enorm, et, quant, oidx, partial);
    vq_finalize<<<1, 256, 0, stream>>>(partial, diffp);
}
